// MinkowskiFeatureFusionBlock_10926396801616
// MI455X (gfx1250) — matrix operand order repaired and checked by reading, not compiled, not run on hardware
//
#include <hip/hip_runtime.h>

// MinkowskiFeatureFusionBlock for MI455X (gfx1250, wave32).
//   x = backbone @ W_top + (text @ W_bot)[batch_idx];  BN(train stats) + ReLU
// Folding the text branch (only B=8 batches) shrinks the GEMM to K=96 and
// removes the gather/concat. Two HBM-bound passes over backbone (stats, then
// apply) using v_wmma_f32_16x16x32_bf16; the backbone tile is DMA'd
// global->LDS by the Tensor Data Mover (TENSORcnt) with HW zero-fill for the
// tail tile and HW LDS padding for bank-conflict-free fragment loads.

#define NCH    96     // CB == CO == 96
#define CTX    128    // CT
#define NBATCH 8
#define TILE_M 64     // rows per block
#define STRAF  108    // fp32 A-tile LDS row pitch (96 + 3 pads of 4 dwords)
#define STRW   104    // bf16 W LDS row pitch (ushorts)
#define EPS_BN 1e-5f

typedef __bf16 bf16_t;
typedef bf16_t v16bf __attribute__((ext_vector_type(16)));
typedef float  v8f   __attribute__((ext_vector_type(8)));
typedef unsigned int u32x4 __attribute__((ext_vector_type(4)));
typedef int i32x8 __attribute__((ext_vector_type(8)));
typedef int i32x4 __attribute__((ext_vector_type(4)));

union Frag16 { v16bf v; unsigned u[8]; bf16_t e[16]; };

#if defined(__HIP_DEVICE_COMPILE__) && __has_builtin(__builtin_amdgcn_tensor_load_to_lds)
#define USE_TDM 1
#else
#define USE_TDM 0
#endif

__device__ __forceinline__ unsigned short f2bf(float f) {
  unsigned u = __float_as_uint(f);              // RNE fp32 -> bf16 (cold path)
  u += 0x7FFFu + ((u >> 16) & 1u);
  return (unsigned short)(u >> 16);
}

// ---------------------------------------------------------------------------
// Kernel 1: zero stats; tc[b][n] = text @ W_bot; prepack W_top^T as bf16
// (padded pitch) so main blocks only dword-copy it from L2-resident scratch.
// ---------------------------------------------------------------------------
__global__ __launch_bounds__(256) void prep_kernel(
    const float* __restrict__ text, const float* __restrict__ W,
    float* __restrict__ tc, float* __restrict__ gsum, float* __restrict__ gsq,
    unsigned short* __restrict__ Wbf)
{
  const int t = threadIdx.x;
  if (t < NCH)          gsum[t] = 0.f;
  else if (t < 2 * NCH) gsq[t - NCH] = 0.f;
  for (int o = t; o < NBATCH * NCH; o += 256) {
    const int b = o / NCH, nn = o % NCH;
    float acc = 0.f;
    for (int k = 0; k < CTX; ++k)
      acc = fmaf(text[b * CTX + k], W[(NCH + k) * NCH + nn], acc);
    tc[o] = acc;
  }
  for (int i = t; i < NCH * NCH; i += 256) {
    const int k = i / NCH, nn = i % NCH;            // coalesced read of W
    Wbf[nn * STRW + k] = f2bf(W[k * NCH + nn]);     // transposed store
  }
}

// ---------------------------------------------------------------------------
// Kernel 3: scale/bias from the accumulated stats
// ---------------------------------------------------------------------------
__global__ __launch_bounds__(128) void finalize_kernel(
    const float* __restrict__ gsum, const float* __restrict__ gsq,
    const float* __restrict__ gamma, const float* __restrict__ beta,
    float* __restrict__ scale, float* __restrict__ bias, int n)
{
  const int t = threadIdx.x;
  if (t < NCH) {
    const float inv_n = 1.0f / (float)n;
    const float mean = gsum[t] * inv_n;
    const float var  = gsq[t] * inv_n - mean * mean;
    const float s    = gamma[t] * rsqrtf(var + EPS_BN);
    scale[t] = s;
    bias[t]  = beta[t] - mean * s;
  }
}

// ---------------------------------------------------------------------------
// Kernels 2 & 4: WMMA GEMM over a 64x96 tile.
// 8 waves = 4 row-strips (16 rows) x 2 column-halves (3 x 16-col WMMA tiles).
// A tile: TDM DMA (fp32, HW-padded pitch STRAF, HW zero-fill past row n).
// ---------------------------------------------------------------------------
template <bool STATS>
__global__ __launch_bounds__(256) void fuse_kernel(
    const float* __restrict__ A, const int* __restrict__ bidx,
    const unsigned* __restrict__ Wsrc,   // prepacked bf16 W_top^T (dwords)
    const float* __restrict__ tc,
    float* __restrict__ gsum, float* __restrict__ gsq,
    const float* __restrict__ scale, const float* __restrict__ bias,
    float* __restrict__ out, int n)
{
  __shared__ float          AldsF[TILE_M * STRAF] __attribute__((aligned(16)));
  __shared__ unsigned short Wlds[NCH * STRW]      __attribute__((aligned(16)));
  __shared__ float tclds[NBATCH * NCH];
  __shared__ int   bix[TILE_M];
  __shared__ float lsum[NCH], lsq[NCH];

  const int tid   = threadIdx.x;
  const int tile0 = blockIdx.x * TILE_M;

#if USE_TDM
  if (tid < 32) {  // wave 0 issues the tile DMA; everyone stages W/tc meanwhile
    const unsigned long long gaddr =
        (unsigned long long)(const void*)A + (unsigned long long)tile0 * (NCH * 4);
    // flat __shared__ pointer: low 32 bits == LDS byte offset (ISA 10.2)
    const unsigned ldsoff = (unsigned)(unsigned long long)(void*)AldsF;
    u32x4 g0;
    g0[0] = 1u;                                   // count=1, load, no gather
    g0[1] = ldsoff;                               // lds_addr
    g0[2] = (unsigned)gaddr;                      // global_addr[31:0]
    g0[3] = (unsigned)((gaddr >> 32) & 0x1FFFFFFu) | (2u << 30);  // [56:32]|type=2
    const unsigned tdim1 = (unsigned)(n - tile0); // HW zero-fills rows >= n
    i32x8 g1;
    g1[0] = (2 << 16)     // data_size = 4B
          | (1 << 20)     // pad_enable
          | (4 << 22)     // pad_interval: every 32 dwords
          | (3 << 25);    // pad_amount: 4 dwords  -> row pitch 108 dwords
    g1[1] = (int)(96u << 16);                       // tensor_dim0 = 96
    g1[2] = (int)((tdim1 & 0xFFFFu) << 16);         // tensor_dim1 lo
    g1[3] = (int)((tdim1 >> 16) | (96u << 16));     // tensor_dim1 hi | tile_dim0
    g1[4] = TILE_M;                                 // tile_dim1 (tile_dim2=0)
    g1[5] = 96;                                     // tensor_dim0_stride
    g1[6] = 0;
    g1[7] = 0;
    i32x4 z4 = {0, 0, 0, 0};
#if defined(__clang_major__) && (__clang_major__ >= 23)
    i32x8 z8 = {0, 0, 0, 0, 0, 0, 0, 0};
    __builtin_amdgcn_tensor_load_to_lds(g0, g1, z4, z4, z8, 0);
#else
    __builtin_amdgcn_tensor_load_to_lds(g0, g1, z4, z4, 0);
#endif
  }
#else
  // Fallback: manual staging into the identical padded fp32 layout.
  for (int i = tid; i < TILE_M * (NCH / 4); i += 256) {
    const int r  = i / (NCH / 4);
    const int c4 = (i % (NCH / 4)) * 4;
    const int row = tile0 + r;
    float4 v = make_float4(0.f, 0.f, 0.f, 0.f);
    if (row < n) v = *reinterpret_cast<const float4*>(&A[(size_t)row * NCH + c4]);
    *reinterpret_cast<float4*>(&AldsF[r * STRAF + c4 + 4 * (c4 >> 5)]) = v;
  }
#endif

  // Stage prepacked W (bf16, already transposed+padded): pure dword copies.
  // Overlaps with the in-flight TDM transfer.
  for (int i = tid; i < (NCH * STRW) / 2; i += 256)
    reinterpret_cast<unsigned*>(Wlds)[i] = Wsrc[i];
  for (int i = tid; i < NBATCH * NCH; i += 256) tclds[i] = tc[i];
  if (tid < TILE_M) {
    const int row = tile0 + tid;
    bix[tid] = (row < n) ? bidx[row] : 0;
  }
  if (STATS) {
    if (tid < NCH)          lsum[tid] = 0.f;
    else if (tid < 2 * NCH) lsq[tid - NCH] = 0.f;
  }
#if USE_TDM
  if (tid < 32) __builtin_amdgcn_s_wait_tensorcnt(0);  // wave 0 joins last
#endif
  __syncthreads();

  const int lane  = tid & 31;
  const int wv    = tid >> 5;
  const int hi    = lane >> 4;        // ISA fragment lane-half
  const int ln    = lane & 15;
  const int mbase = (wv >> 1) * 16;   // row strip
  const int ch    = wv & 1;           // column half (3 tiles of 16 cols)

  v8f acc[3];
  #pragma unroll
  for (int t = 0; t < 3; ++t) { v8f z = {}; acc[t] = z; }

  #pragma unroll
  for (int ks = 0; ks < 3; ++ks) {
    const int kb = ks * 32;
    // A frag 16x32: lanes 0-15 -> K {kb..kb+7, kb+16..23}; lanes 16-31 -> +8.
    // Padded pitch: dword addr = row*108 + k + 4*(k/32) -> base row*108+36*ks+8*hi
    Frag16 a;
    {
      const float* ar = &AldsF[(mbase + ln) * STRAF + 36 * ks + 8 * hi];
      const float4 f0 = *reinterpret_cast<const float4*>(ar + 0);
      const float4 f1 = *reinterpret_cast<const float4*>(ar + 4);
      const float4 f2 = *reinterpret_cast<const float4*>(ar + 16);
      const float4 f3 = *reinterpret_cast<const float4*>(ar + 20);
      // Let the compiler lower f32->bf16 (FP_ROUND) -> native v_cvt_*bf16*
      a.e[0]  = (bf16_t)f0.x; a.e[1]  = (bf16_t)f0.y;
      a.e[2]  = (bf16_t)f0.z; a.e[3]  = (bf16_t)f0.w;
      a.e[4]  = (bf16_t)f1.x; a.e[5]  = (bf16_t)f1.y;
      a.e[6]  = (bf16_t)f1.z; a.e[7]  = (bf16_t)f1.w;
      a.e[8]  = (bf16_t)f2.x; a.e[9]  = (bf16_t)f2.y;
      a.e[10] = (bf16_t)f2.z; a.e[11] = (bf16_t)f2.w;
      a.e[12] = (bf16_t)f3.x; a.e[13] = (bf16_t)f3.y;
      a.e[14] = (bf16_t)f3.z; a.e[15] = (bf16_t)f3.w;
    }
    #pragma unroll
    for (int tt = 0; tt < 3; ++tt) {
      const int t = ch * 3 + tt;
      // B frag 32x16: col = 16t+ln; lanes 0-15 -> K {kb..kb+7, kb+16..23}; 16-31 -> +8
      Frag16 b;
      const unsigned short* br = &Wlds[(t * 16 + ln) * STRW + kb + 8 * hi];
      #pragma unroll
      for (int j = 0; j < 8; ++j)
        b.u[j] = *reinterpret_cast<const unsigned*>(br + 2 * j + (j >= 4 ? 8 : 0));
      acc[tt] = __builtin_amdgcn_wmma_f32_16x16x32_bf16(
          false, a.v, false, b.v, (short)0, acc[tt], false, false);
    }
  }

  // Epilogue: C/D layout -> (M = r + 8*hi, N = ln) per VGPR r.
  #pragma unroll
  for (int tt = 0; tt < 3; ++tt) {
    const int ncol = (ch * 3 + tt) * 16 + ln;
    float s = 0.f, q = 0.f;
    float sc = 0.f, bs = 0.f;
    if (!STATS) { sc = scale[ncol]; bs = bias[ncol]; }
    #pragma unroll
    for (int r = 0; r < 8; ++r) {
      const int mloc = mbase + r + (hi ? 8 : 0);
      const int row  = tile0 + mloc;
      const bool valid = row < n;
      const float x = acc[tt][r] + (valid ? tclds[bix[mloc] * NCH + ncol] : 0.f);
      if (STATS) {
        const float xv = valid ? x : 0.f;   // padded rows contribute exact 0
        s += xv; q += xv * xv;
      } else if (valid) {
        out[(size_t)row * NCH + ncol] = fmaxf(fmaf(x, sc, bs), 0.f);
      }
    }
    if (STATS) {
      s += __shfl_xor(s, 16, 32);   // pair lanes sharing a column (wave32)
      q += __shfl_xor(q, 16, 32);
      if (hi == 0) { atomicAdd(&lsum[ncol], s); atomicAdd(&lsq[ncol], q); }
    }
  }
  if (STATS) {
    __syncthreads();
    for (int i = tid; i < NCH; i += 256) {
      atomicAdd(&gsum[i], lsum[i]);
      atomicAdd(&gsq[i],  lsq[i]);
    }
  }
}

// ---------------------------------------------------------------------------
extern "C" void kernel_launch(void* const* d_in, const int* in_sizes, int n_in,
                              void* d_out, int out_size, void* d_ws, size_t ws_size,
                              hipStream_t stream) {
  const float* backbone = (const float*)d_in[0];   // [N, 96]
  const int*   batch    = (const int*)  d_in[1];   // [N]
  const float* text     = (const float*)d_in[2];   // [8, 128]
  const float* W        = (const float*)d_in[3];   // [224, 96]
  const float* gamma    = (const float*)d_in[4];   // [96]
  const float* beta     = (const float*)d_in[5];   // [96]
  float* out = (float*)d_out;                      // [N, 96]
  const int n = in_sizes[0] / NCH;

  float* ws    = (float*)d_ws;
  float* tc    = ws;                 // 768 floats
  float* gsum  = ws + 768;           // 96
  float* gsq   = ws + 864;           // 96
  float* scale = ws + 960;           // 96
  float* bias  = ws + 1056;          // 96
  unsigned short* Wbf = (unsigned short*)(ws + 1152);   // 96*104 ushorts
  const unsigned* Wsrc = (const unsigned*)Wbf;

  const int grid = (n + TILE_M - 1) / TILE_M;
  prep_kernel<<<1, 256, 0, stream>>>(text, W, tc, gsum, gsq, Wbf);
  fuse_kernel<true><<<grid, 256, 0, stream>>>(backbone, batch, Wsrc, tc, gsum, gsq,
                                              nullptr, nullptr, nullptr, n);
  finalize_kernel<<<1, 128, 0, stream>>>(gsum, gsq, gamma, beta, scale, bias, n);
  fuse_kernel<false><<<grid, 256, 0, stream>>>(backbone, batch, Wsrc, tc, gsum, gsq,
                                               scale, bias, out, n);
}
